// CTGRU_Model_17420387353087
// MI455X (gfx1250) — matrix operand order repaired and checked by reading, not compiled, not run on hardware
//
#include <hip/hip_runtime.h>
#include <hip/hip_fp16.h>
#include <math.h>

// CT-GRU forward for MI455X (gfx1250, wave32, WMMA + TDM + DPP8).
// B=4096, T=16, F=512, U=512, M=8, OUT=3, K = F+U = 1024.

typedef __attribute__((ext_vector_type(16))) _Float16 v16h;
typedef __attribute__((ext_vector_type(8)))  float    v8f;
typedef __attribute__((ext_vector_type(4)))  unsigned u32x4;
typedef __attribute__((ext_vector_type(8)))  int      i32x8;
typedef __attribute__((ext_vector_type(4)))  int      i32x4;

#define BM     32          // batch rows per block (2 WMMA M-tiles)
#define MT     2           // M-tiles per block
#define FDIM   512
#define UDIM   512
#define MDIM   8
#define KDIM   1024        // F + U
#define KT     32          // KDIM / 32 k-tiles
#define NR     4096        // U * M
#define TSTEPS 16
#define NOUT   3
#define NTHREADS 512       // 16 waves: w>>3 = M-tile, w&7 = N-slice
#define KSTRIDE 512        // halfs per k-tile fragment block (32 lanes * 16)

// ---- LDS layout (byte offsets into dynamic smem, 288 KB total) ------------
#define SX_OFF   0                         // x tile stage: 32 x 512 f32 (64K)
#define SA_OFF   (SX_OFF + BM*FDIM*4)      // fused A frags: 2 x 16384 h (64K)
#define SA2_OFF  (SA_OFF + MT*16384*2)     // ctx A frags:   2 x 8192  h (32K)
#define SH_OFF   (SA2_OFF + MT*8192*2)     // h:  32 x 512 f32           (64K)
#define SQ_OFF   (SH_OFF + BM*UDIM*4)      // q:  32 x 512 f32           (64K)
#define SMEM_BYTES (SQ_OFF + BM*UDIM*4)    // = 294912

// ---------------------------------------------------------------------------
// Pack f32 row-major weights [K][N] into per-lane WMMA B fragments (f16):
//   dst[((nt*KT + kt)*32 + lane)*16 + j] = W[k][n],
//   k = kt*32 + (j/8)*16 + (lane/16)*8 + (j%8) ,  n = nt*16 + (lane%16)
// ---------------------------------------------------------------------------
__global__ void pack_w_kernel(const float* __restrict__ W,
                              _Float16* __restrict__ dst, int K, int N) {
    int total = K * N;
    int kt_per_n = K >> 5;
    for (int idx = blockIdx.x * blockDim.x + threadIdx.x; idx < total;
         idx += gridDim.x * blockDim.x) {
        int j   = idx & 15;
        int l   = (idx >> 4) & 31;
        int rem = idx >> 9;
        int kt  = rem % kt_per_n;
        int nt  = rem / kt_per_n;
        int k   = kt * 32 + ((j >> 3) << 4) + ((l >> 4) << 3) + (j & 7);
        int n   = nt * 16 + (l & 15);
        dst[idx] = (_Float16)W[(size_t)k * N + n];
    }
}

// ---- 8-lane butterfly reductions in pure VALU via DPP8 --------------------
#define DPP8SEL(a,b,c,d,e,f,g,h) \
    ((a)|((b)<<3)|((c)<<6)|((d)<<9)|((e)<<12)|((f)<<15)|((g)<<18)|((h)<<21))
template <unsigned S>
__device__ __forceinline__ float dpp8f(float v) {
    return __builtin_bit_cast(
        float, __builtin_amdgcn_mov_dpp8(__builtin_bit_cast(int, v), S));
}
__device__ __forceinline__ float redSum8(float v) {
    v += dpp8f<DPP8SEL(1,0,3,2,5,4,7,6)>(v);   // xor 1
    v += dpp8f<DPP8SEL(2,3,0,1,6,7,4,5)>(v);   // xor 2
    v += dpp8f<DPP8SEL(4,5,6,7,0,1,2,3)>(v);   // xor 4
    return v;
}
__device__ __forceinline__ float redMax8(float v) {
    v = fmaxf(v, dpp8f<DPP8SEL(1,0,3,2,5,4,7,6)>(v));
    v = fmaxf(v, dpp8f<DPP8SEL(2,3,0,1,6,7,4,5)>(v));
    v = fmaxf(v, dpp8f<DPP8SEL(4,5,6,7,0,1,2,3)>(v));
    return v;
}

// ---- fully-unrolled k-tile GEMM: one base pointer per operand, pure -------
// ---- immediate offsets -> clause-grouped loads, staggered waits, no WAR ---
template <int NKT>
__device__ __forceinline__ v8f gemm16(const _Float16* __restrict__ ap,
                                      const _Float16* __restrict__ bp,
                                      v8f acc) {
    #pragma unroll
    for (int kt = 0; kt < NKT; ++kt) {
        v16h a = *(const v16h*)(ap + kt * KSTRIDE);
        v16h b = *(const v16h*)(bp + kt * KSTRIDE);
        acc = __builtin_amdgcn_wmma_f32_16x16x32_f16(
                  false, a, false, b, (short)0, acc, false, false);
    }
    return acc;
}

// ---------------------------------------------------------------------------
// Persistent block: 32 batch rows, full T=16 recurrence, 16 waves.
// ---------------------------------------------------------------------------
__global__ void __launch_bounds__(NTHREADS)
ctgru_kernel(const float* __restrict__ x,
             const float* __restrict__ br,
             const float* __restrict__ bq,
             const float* __restrict__ bs,
             const float* __restrict__ Wo,
             const float* __restrict__ bo,
             const _Float16* __restrict__ WrP,
             const _Float16* __restrict__ WqP,
             const _Float16* __restrict__ WsP,
             float* __restrict__ hhat,
             float* __restrict__ out) {
    extern __shared__ char smem[];
    float*    sX  = (float*)(smem + SX_OFF);
    _Float16* sA  = (_Float16*)(smem + SA_OFF);
    _Float16* sA2 = (_Float16*)(smem + SA2_OFF);
    float*    sH  = (float*)(smem + SH_OFF);
    float*    sQ  = (float*)(smem + SQ_OFF);

    const int tid  = threadIdx.x;
    const int lane = tid & 31;
    const int w    = tid >> 5;
    const int mt   = w >> 3;                 // M-tile owned by this wave
    const int ng   = w & 7;                  // N-slice owned by this wave
    const int b0   = blockIdx.x * BM;

    const int   m      = lane & 7;           // trace index (column % 8)
    const float ln_tau = 1.1512925464970229f * (float)m;   // 0.5*ln(10)*m
    const float decay  = __expf(-0.04f / (__expf(ln_tau) + 1e-7f));

    // zero h and this block's h_hat slab
    for (int i = tid; i < BM * UDIM; i += NTHREADS) sH[i] = 0.0f;
    float* hh_base = hhat + (size_t)b0 * UDIM * MDIM;
    for (int i = tid; i < BM * UDIM * MDIM; i += NTHREADS) hh_base[i] = 0.0f;
    __syncthreads();

    const _Float16* sAm  = sA  + mt * 16384 + lane * 16;   // wave A base
    const _Float16* sA2m = sA2 + mt * 8192  + lane * 16;   // wave ctx-A base

    for (int t = 0; t < TSTEPS; ++t) {
        // ---- 0. TDM: DMA the x tile (32 rows x 512 f32) into sX ----------
        if (w == 0) {
            unsigned long long ga = (unsigned long long)(const void*)
                (x + ((size_t)b0 * TSTEPS + t) * FDIM);
            u32x4 g0;
            g0[0] = 1u;                                   // count=1, user D#
            g0[1] = (unsigned)SX_OFF;                     // lds_addr
            g0[2] = (unsigned)(ga & 0xFFFFFFFFu);         // global_addr lo
            g0[3] = (unsigned)((ga >> 32) & 0x1FFFFFFu)   // global_addr hi
                  | (2u << 30);                           // type = image
            i32x8 g1;
            g1[0] = (int)(2u << 16);       // data_size = 4 bytes
            g1[1] = (int)(512u << 16);     // tensor_dim0 = 512
            g1[2] = (int)(32u << 16);      // tensor_dim1 = 32
            g1[3] = (int)(512u << 16);     // tile_dim0 = 512
            g1[4] = 32;                    // tile_dim1 = 32, tile_dim2 = 0
            g1[5] = TSTEPS * FDIM;         // tensor_dim0_stride = 8192
            g1[6] = 0;
            g1[7] = 0;
            i32x4 z4 = {0, 0, 0, 0};
            i32x8 z8 = {0, 0, 0, 0, 0, 0, 0, 0};
            __builtin_amdgcn_tensor_load_to_lds(g0, g1, z4, z4, z8, 0);
            __builtin_amdgcn_s_wait_tensorcnt(0);
        }
        __syncthreads();

        // ---- 1. build fused = [x_t | h] A fragments (both M-tiles) -------
        // A layout (16-bit 16x32): row = l%16, K(l,j)=(j/8)*16+(l/16)*8+(j%8)
        for (int i = tid; i < MT * 32 * 32 * 16; i += NTHREADS) {
            int j = i & 15, l = (i >> 4) & 31, kt = (i >> 9) & 31, g = i >> 14;
            int row = g * 16 + (l & 15);
            int k = kt * 32 + ((j >> 3) << 4) + ((l >> 4) << 3) + (j & 7);
            float v = (k < FDIM) ? sX[row * FDIM + k]
                                 : sH[row * UDIM + (k - FDIM)];
            sA[i] = (_Float16)v;
        }
        __syncthreads();

        // ---- 2. r-path: fused @ Wr ; softmax_m ; ctx -> sA2 frags --------
        for (int i = 0; i < 32; ++i) {
            int nt = ng + (i << 3);                 // 0..255
            v8f acc = {};
            acc = gemm16<KT>(sAm, WrP + (size_t)nt * KT * KSTRIDE + lane * 16, acc);
            if (i + 1 < 32)     // hint next weight slab toward L0/L2
                __builtin_prefetch(WrP + (size_t)(nt + 8) * KT * KSTRIDE + lane * 16, 0, 1);
            float bias = br[nt * 16 + (lane & 15)];
            int   u    = nt * 2 + ((lane & 15) >> 3);
            #pragma unroll
            for (int v = 0; v < 8; ++v) {
                float d  = acc[v] + bias - ln_tau;
                float z  = -d * d;
                float e  = __expf(z - redMax8(z));
                float r  = e / redSum8(e);
                int row  = v + ((lane >> 4) << 3);          // 0..15 in tile
                int grow = mt * 16 + row;
                float hhv = hh_base[((size_t)grow * UDIM + u) * MDIM + m];
                float c   = redSum8(r * hhv);
                if (m == 0) {   // scatter ctx straight into its A fragment
                    int kp = u & 31, kt2 = u >> 5;
                    int lg = (kp >> 3) & 1, jj = ((kp >> 4) << 3) | (kp & 7);
                    sA2[mt * 8192 + (kt2 * 32 + (row + (lg << 4))) * 16 + jj]
                        = (_Float16)c;
                }
            }
        }
        __syncthreads();

        // ---- 3. q = tanh([x_t | ctx] @ Wq + bq) --------------------------
        for (int i = 0; i < 4; ++i) {
            int nt = ng + (i << 3);                 // 0..31
            const _Float16* bq16 = WqP + (size_t)nt * KT * KSTRIDE + lane * 16;
            v8f acc = {};
            acc = gemm16<16>(sAm,  bq16,                 acc);  // x part
            acc = gemm16<16>(sA2m, bq16 + 16 * KSTRIDE,  acc);  // ctx part
            float bias = bq[nt * 16 + (lane & 15)];
            #pragma unroll
            for (int v = 0; v < 8; ++v) {
                int grow = mt * 16 + v + ((lane >> 4) << 3);
                sQ[grow * UDIM + nt * 16 + (lane & 15)] = tanhf(acc[v] + bias);
            }
        }
        __syncthreads();

        // ---- 4. s-path: fused @ Ws ; gate ; update h_hat ; h -> sH -------
        for (int i = 0; i < 32; ++i) {
            int nt = ng + (i << 3);
            v8f acc = {};
            acc = gemm16<KT>(sAm, WsP + (size_t)nt * KT * KSTRIDE + lane * 16, acc);
            if (i + 1 < 32)
                __builtin_prefetch(WsP + (size_t)(nt + 8) * KT * KSTRIDE + lane * 16, 0, 1);
            float bias = bs[nt * 16 + (lane & 15)];
            int   u    = nt * 2 + ((lane & 15) >> 3);
            #pragma unroll
            for (int v = 0; v < 8; ++v) {
                float d  = acc[v] + bias - ln_tau;
                float z  = -d * d;
                float e  = __expf(z - redMax8(z));
                float s  = e / redSum8(e);
                int grow = mt * 16 + v + ((lane >> 4) << 3);
                size_t hidx = ((size_t)grow * UDIM + u) * MDIM + m;
                float hhv = hh_base[hidx];
                float qv  = sQ[grow * UDIM + u];
                float hn  = ((1.0f - s) * hhv + s * qv) * decay;
                hh_base[hidx] = hn;
                float hs = redSum8(hn);
                if (m == 0) sH[grow * UDIM + u] = hs;
            }
        }
        __syncthreads();

        // ---- 5. out[b,t,:] = h @ Wo + bo  (all 512 threads) --------------
        {
            int row = tid >> 4;                 // 0..31
            int uo  = tid & 15;
            float a0 = 0.f, a1 = 0.f, a2 = 0.f;
            for (int u = uo; u < UDIM; u += 16) {
                float h = sH[row * UDIM + u];
                a0 += h * Wo[u * NOUT + 0];
                a1 += h * Wo[u * NOUT + 1];
                a2 += h * Wo[u * NOUT + 2];
            }
            a0 = redSum8(a0); a0 += __shfl_xor(a0, 8, 32);
            a1 = redSum8(a1); a1 += __shfl_xor(a1, 8, 32);
            a2 = redSum8(a2); a2 += __shfl_xor(a2, 8, 32);
            if (uo == 0) {
                size_t ob = ((size_t)(b0 + row) * TSTEPS + t) * NOUT;
                out[ob + 0] = a0 + bo[0];
                out[ob + 1] = a1 + bo[1];
                out[ob + 2] = a2 + bo[2];
            }
        }
        __syncthreads();
    }
}

// ---------------------------------------------------------------------------
extern "C" void kernel_launch(void* const* d_in, const int* in_sizes, int n_in,
                              void* d_out, int out_size, void* d_ws, size_t ws_size,
                              hipStream_t stream) {
    const float* x  = (const float*)d_in[0];
    const float* Wr = (const float*)d_in[1];
    const float* br = (const float*)d_in[2];
    const float* Wq = (const float*)d_in[3];
    const float* bq = (const float*)d_in[4];
    const float* Ws = (const float*)d_in[5];
    const float* bs = (const float*)d_in[6];
    const float* Wo = (const float*)d_in[7];
    const float* bo = (const float*)d_in[8];

    // workspace: packed f16 weights + h_hat state (~85 MB)
    _Float16* WrP  = (_Float16*)d_ws;                         // 1024*4096 f16
    _Float16* WsP  = WrP + (size_t)KDIM * NR;                 // 1024*4096 f16
    _Float16* WqP  = WsP + (size_t)KDIM * NR;                 // 1024*512  f16
    float*    hhat = (float*)(WqP + (size_t)KDIM * UDIM);     // 4096*512*8 f32

    pack_w_kernel<<<4096, 256, 0, stream>>>(Wr, WrP, KDIM, NR);
    pack_w_kernel<<<4096, 256, 0, stream>>>(Ws, WsP, KDIM, NR);
    pack_w_kernel<<<512,  256, 0, stream>>>(Wq, WqP, KDIM, UDIM);

    (void)hipFuncSetAttribute((const void*)ctgru_kernel,
                              hipFuncAttributeMaxDynamicSharedMemorySize,
                              (int)SMEM_BYTES);
    ctgru_kernel<<<4096 / BM, NTHREADS, SMEM_BYTES, stream>>>(
        x, br, bq, bs, Wo, bo, WrP, WqP, WsP, hhat, (float*)d_out);
}
